// SCM_MLP_49460843381367
// MI455X (gfx1250) — compile-verified, matrix operand order checked by reading, not run on hardware
//
#include <hip/hip_runtime.h>
#include <hip/hip_bf16.h>

// ---------------------------------------------------------------------------
// SCM_MLP on MI455X (gfx1250): 4 chained masked GEMMs (4096^3) with fused
// lognormal-noise + ReLU epilogue.
//  - bf16 WMMA (v_wmma_f32_16x16x32_bf16), wave32, 8 waves/block, 128x128x64
//  - tile staging via GLOBAL_LOAD_ASYNC_TO_LDS_B128 (ASYNCcnt) with LDS
//    double buffering: DMA overlapped with WMMA, no VGPR staging.
// ---------------------------------------------------------------------------

typedef __attribute__((ext_vector_type(8)))  __bf16 v8bf;
typedef __attribute__((ext_vector_type(16))) __bf16 v16bf;
typedef __attribute__((ext_vector_type(8)))  float  v8f;
typedef __attribute__((ext_vector_type(4)))  int    v4i;

#define GLOBAL_AS __attribute__((address_space(1)))
#define LDS_AS    __attribute__((address_space(3)))

typedef GLOBAL_AS v4i* gv4ip;   // global int4*  (async-DMA source)
typedef LDS_AS    v4i* lv4ip;   // LDS int4*     (async-DMA destination)

#define H_DIM 4096
#define B_DIM 4096
#define L_DIM 4
#define OSTRIDE (L_DIM * H_DIM)   // 16384

#define BM 128
#define BN 128
#define BK 64
#define LDS_STRIDE 72             // 64 cols + 8 bf16 pad -> 144B rows, conflict-free b128 reads
#define NK (H_DIM / BK)           // 64

// ---- elementwise: masked weight -> bf16 -----------------------------------
__global__ __launch_bounds__(256) void k_maskw_bf16(const float* __restrict__ w,
                                                    const float* __restrict__ m,
                                                    __bf16* __restrict__ dst) {
    size_t i = ((size_t)blockIdx.x * 256 + threadIdx.x) * 4;
    float4 wv = *(const float4*)(w + i);
    float4 mv = *(const float4*)(m + i);
    dst[i + 0] = (__bf16)(wv.x * mv.x);
    dst[i + 1] = (__bf16)(wv.y * mv.y);
    dst[i + 2] = (__bf16)(wv.z * mv.z);
    dst[i + 3] = (__bf16)(wv.w * mv.w);
}

// ---- elementwise: fp32 -> bf16 (contiguous) -------------------------------
__global__ __launch_bounds__(256) void k_f32_to_bf16(const float* __restrict__ src,
                                                     __bf16* __restrict__ dst) {
    size_t i = ((size_t)blockIdx.x * 256 + threadIdx.x) * 4;
    float4 v = *(const float4*)(src + i);
    dst[i + 0] = (__bf16)v.x;
    dst[i + 1] = (__bf16)v.y;
    dst[i + 2] = (__bf16)v.z;
    dst[i + 3] = (__bf16)v.w;
}

// ---- strided: activation column block of d_out -> bf16 A operand ----------
__global__ __launch_bounds__(256) void k_act_to_bf16(const float* __restrict__ out,
                                                     int colOff,
                                                     __bf16* __restrict__ dst) {
    size_t i = ((size_t)blockIdx.x * 256 + threadIdx.x) * 4;   // index into B*H
    size_t b = i >> 12;            // / H_DIM
    size_t c = i & (H_DIM - 1);    // % H_DIM (vector never crosses a row: H%4==0)
    float4 v = *(const float4*)(out + b * OSTRIDE + colOff + c);
    dst[i + 0] = (__bf16)v.x;
    dst[i + 1] = (__bf16)v.y;
    dst[i + 2] = (__bf16)v.z;
    dst[i + 3] = (__bf16)v.w;
}

// ---- main GEMM + noise + relu ---------------------------------------------
// out[b, oc] = relu( sum_k A[b,k]*W[oc,k] + exp(mu[oc] + sigma[oc]*z[b,oc]) )
__global__ __launch_bounds__(256) void k_gemm_noise_relu(
    const __bf16* __restrict__ A,     // [B_DIM, H_DIM] bf16
    const __bf16* __restrict__ W,     // [H_DIM, H_DIM] bf16 (masked)
    const float*  __restrict__ zl,    // [B_DIM, H_DIM]
    const float*  __restrict__ mul,   // [H_DIM]
    const float*  __restrict__ sgl,   // [H_DIM]
    float* __restrict__ outcol)       // d_out + l*H_DIM, row stride OSTRIDE
{
    __shared__ __bf16 ldsA[2][BM * LDS_STRIDE];
    __shared__ __bf16 ldsW[2][BN * LDS_STRIDE];

    const int tid   = threadIdx.x;
    const int lane  = tid & 31;
    const int wave  = tid >> 5;          // 0..7
    const int wm    = wave >> 2;         // 0..1 : 64-row strip
    const int wn    = wave & 3;          // 0..3 : 32-col strip
    const int g     = lane >> 4;         // half-wave select (K-interleave)
    const int ln    = lane & 15;

    const int n0 = blockIdx.x * BN;      // output-feature base
    const int m0 = blockIdx.y * BM;      // batch-row base

    v8f acc[4][2];
    const v8f vzero = {0.f, 0.f, 0.f, 0.f, 0.f, 0.f, 0.f, 0.f};
#pragma unroll
    for (int mi = 0; mi < 4; ++mi)
#pragma unroll
        for (int nj = 0; nj < 2; ++nj) acc[mi][nj] = vzero;

    // per-thread tile-chunk geometry (fixed across K-steps)
    // 128 rows x 64 bf16 cols = 1024 chunks of 16B; 4 chunks per thread.
    int rowc[4], colc[4];
#pragma unroll
    for (int i = 0; i < 4; ++i) {
        int chunk = tid + i * 256;
        rowc[i] = chunk >> 3;
        colc[i] = (chunk & 7) * 8;
    }

    // issue one tile's async DMA (8 x B128 per thread, tracked by ASYNCcnt)
    auto issue_tile = [&](int buf, int kt) {
        const int k0 = kt * BK;
#pragma unroll
        for (int i = 0; i < 4; ++i) {
            __builtin_amdgcn_global_load_async_to_lds_b128(
                (gv4ip)(A + (size_t)(m0 + rowc[i]) * H_DIM + k0 + colc[i]),
                (lv4ip)&ldsA[buf][rowc[i] * LDS_STRIDE + colc[i]],
                0, 0);
            __builtin_amdgcn_global_load_async_to_lds_b128(
                (gv4ip)(W + (size_t)(n0 + rowc[i]) * H_DIM + k0 + colc[i]),
                (lv4ip)&ldsW[buf][rowc[i] * LDS_STRIDE + colc[i]],
                0, 0);
        }
    };

    issue_tile(0, 0);

    for (int kt = 0; kt < NK; ++kt) {
        const int cur = kt & 1;
        // prefetch next tile into the other buffer; async copies overlap WMMA
        if (kt + 1 < NK) {
            issue_tile(cur ^ 1, kt + 1);
            __builtin_amdgcn_s_wait_asynccnt(8);   // cur tile resident, next in flight
        } else {
            __builtin_amdgcn_s_wait_asynccnt(0);
        }
        __syncthreads();                            // publish cur tile to all waves

        // two K=32 WMMA steps over the BK=64 tile
#pragma unroll
        for (int kk = 0; kk < 2; ++kk) {
            v16bf af[4], bfr[2];
#pragma unroll
            for (int mi = 0; mi < 4; ++mi) {
                const __bf16* p = &ldsA[cur][(wm * 64 + mi * 16 + ln) * LDS_STRIDE + kk * 32 + g * 8];
                v8bf lo = *(const v8bf*)p;
                v8bf hi = *(const v8bf*)(p + 16);
                af[mi] = __builtin_shufflevector(lo, hi, 0,1,2,3,4,5,6,7,8,9,10,11,12,13,14,15);
            }
#pragma unroll
            for (int nj = 0; nj < 2; ++nj) {
                const __bf16* p = &ldsW[cur][(wn * 32 + nj * 16 + ln) * LDS_STRIDE + kk * 32 + g * 8];
                v8bf lo = *(const v8bf*)p;
                v8bf hi = *(const v8bf*)(p + 16);
                bfr[nj] = __builtin_shufflevector(lo, hi, 0,1,2,3,4,5,6,7,8,9,10,11,12,13,14,15);
            }
#pragma unroll
            for (int mi = 0; mi < 4; ++mi)
#pragma unroll
                for (int nj = 0; nj < 2; ++nj)
                    acc[mi][nj] = __builtin_amdgcn_wmma_f32_16x16x32_bf16(
                        false, af[mi], false, bfr[nj], (short)0, acc[mi][nj], false, false);
        }
        __syncthreads();   // protect the buffer that next iteration's DMA re-targets
    }

    // epilogue: + exp(mu + sigma*z), relu, store fp32 into concat column block
#pragma unroll
    for (int nj = 0; nj < 2; ++nj) {
        const int oc = n0 + wn * 32 + nj * 16 + ln;   // feature index in [0,H)
        const float muv = mul[oc];
        const float sgv = sgl[oc];
#pragma unroll
        for (int mi = 0; mi < 4; ++mi) {
            v8f c = acc[mi][nj];
#pragma unroll
            for (int vi = 0; vi < 8; ++vi) {
                const int rm = m0 + wm * 64 + mi * 16 + g * 8 + vi;  // batch row
                const float zv = zl[(size_t)rm * H_DIM + oc];
                float val = c[vi] + __expf(fmaf(sgv, zv, muv));
                val = fmaxf(val, 0.0f);
                outcol[(size_t)rm * OSTRIDE + oc] = val;
            }
        }
    }
}

// ---------------------------------------------------------------------------
extern "C" void kernel_launch(void* const* d_in, const int* in_sizes, int n_in,
                              void* d_out, int out_size, void* d_ws, size_t ws_size,
                              hipStream_t stream) {
    (void)in_sizes; (void)n_in; (void)out_size; (void)ws_size;

    const float* x  = (const float*)d_in[0];   // [B,H]
    const float* w  = (const float*)d_in[1];   // [L,H,H]
    const float* mk = (const float*)d_in[2];   // [L,H,H]
    const float* mu = (const float*)d_in[3];   // [L,H]
    const float* sg = (const float*)d_in[4];   // [L,H]
    const float* z  = (const float*)d_in[5];   // [L,B,H]
    float* out      = (float*)d_out;           // [B, L*H]

    const size_t BH = (size_t)B_DIM * H_DIM;   // 16,777,216
    const size_t HH = (size_t)H_DIM * H_DIM;   // 16,777,216

    __bf16* wsA = (__bf16*)d_ws;               // 32 MB: bf16 activations (GEMM A)
    __bf16* wsW = wsA + BH;                    // 32 MB: bf16 masked weights

    const int cvtBlocks = (int)(BH / (256 * 4));          // 16384
    dim3 gemmGrid(H_DIM / BN, B_DIM / BM);                // (32, 32)

    // x -> bf16 A operand
    k_f32_to_bf16<<<cvtBlocks, 256, 0, stream>>>(x, wsA);

    for (int l = 0; l < L_DIM; ++l) {
        // masked weights for this layer -> bf16
        k_maskw_bf16<<<cvtBlocks, 256, 0, stream>>>(w + (size_t)l * HH,
                                                    mk + (size_t)l * HH, wsW);
        // GEMM + lognormal noise + relu, writing column block l of d_out
        k_gemm_noise_relu<<<gemmGrid, 256, 0, stream>>>(
            wsA, wsW,
            z + (size_t)l * BH,
            mu + (size_t)l * H_DIM,
            sg + (size_t)l * H_DIM,
            out + (size_t)l * H_DIM);
        // re-materialize this layer's activation as bf16 input for next layer
        if (l + 1 < L_DIM)
            k_act_to_bf16<<<cvtBlocks, 256, 0, stream>>>(out, l * H_DIM, wsA);
    }
}
